// LesionROIExtractor_52974126629494
// MI455X (gfx1250) — matrix operand order repaired and checked by reading, not compiled, not run on hardware
//
#include <hip/hip_runtime.h>
#include <hip/hip_bf16.h>

typedef __attribute__((ext_vector_type(2)))  __bf16 v2bf;
typedef __attribute__((ext_vector_type(8)))  __bf16 v8bf;
typedef __attribute__((ext_vector_type(16))) __bf16 v16bf;
typedef __attribute__((ext_vector_type(8)))  float  v8f;

#define ROI   7
#define SR    2
#define FH    64
#define FW    64
#define FC    768
#define NBOX  256
#define K1    (ROI * ROI * FC)   /* 37632 */
#define N1    1024
#define N2    512

// ---------------------------------------------------------------------------
// RoIAlign (torchvision, aligned=false, sampling_ratio=2) fused with bf16
// pack.  A is [N, K1] with k = bin*768 + c (channel-minor) so the channel-last
// gathers give contiguous coalesced stores; the GEMM un-permutes on the
// W1-row side (W1 rows are contiguous 4KB runs, so the permutation is free).
// ---------------------------------------------------------------------------
__device__ __forceinline__ void interp_axis(float t, int size, int& lo, int& hi,
                                            float& fr, bool& valid) {
  valid = (t > -1.0f) && (t < (float)size);
  t = fmaxf(t, 0.0f);
  lo = min((int)floorf(t), size - 1);
  hi = min(lo + 1, size - 1);
  if (lo >= size - 1) t = (float)lo;
  fr = t - (float)lo;
}

__global__ __launch_bounds__(256) void roi_align_pack_kernel(
    const float* __restrict__ fmap,    // [B,H,W,C] channel-last f32
    const float* __restrict__ boxes,   // [N,5]
    __bf16* __restrict__ A) {          // [N, K1] bf16, k = bin*FC + c
  const int bin = blockIdx.x;          // 0..48
  const int n   = blockIdx.y;          // 0..255
  const int ph = bin / ROI, pw = bin % ROI;

  const float* bx = boxes + n * 5;
  const int   b  = (int)bx[0];
  const float x1 = bx[1], y1 = bx[2], x2 = bx[3], y2 = bx[4];
  const float bin_w = fmaxf(x2 - x1, 1.0f) * (1.0f / ROI);
  const float bin_h = fmaxf(y2 - y1, 1.0f) * (1.0f / ROI);

  int   off[16];
  float wt[16];
#pragma unroll
  for (int iy = 0; iy < SR; ++iy) {
    const float gy = (float)ph + ((float)iy + 0.5f) * (1.0f / SR);
    const float yc = y1 + gy * bin_h;
    int yl, yh; float ly; bool vy;
    interp_axis(yc, FH, yl, yh, ly, vy);
    const float hy = 1.0f - ly;
#pragma unroll
    for (int ix = 0; ix < SR; ++ix) {
      const float gx = (float)pw + ((float)ix + 0.5f) * (1.0f / SR);
      const float xc = x1 + gx * bin_w;
      int xl, xh; float lx; bool vx;
      interp_axis(xc, FW, xl, xh, lx, vx);
      const float hx = 1.0f - lx;
      const float vm = (vy && vx) ? 0.25f : 0.0f;   // validity * 1/(SR*SR)
      const int s = (iy * SR + ix) * 4;
      off[s + 0] = ((b * FH + yl) * FW + xl) * FC;  wt[s + 0] = vm * hy * hx;
      off[s + 1] = ((b * FH + yl) * FW + xh) * FC;  wt[s + 1] = vm * hy * lx;
      off[s + 2] = ((b * FH + yh) * FW + xl) * FC;  wt[s + 2] = vm * ly * hx;
      off[s + 3] = ((b * FH + yh) * FW + xh) * FC;  wt[s + 3] = vm * ly * lx;
    }
  }

  __bf16* dst = A + (size_t)n * K1 + (size_t)bin * FC;
  for (int c = threadIdx.x; c < FC; c += blockDim.x) {
    float acc = 0.0f;
#pragma unroll
    for (int s = 0; s < 16; ++s) acc += wt[s] * fmap[off[s] + c];
    dst[c] = (__bf16)acc;
  }
}

// ---------------------------------------------------------------------------
// out = act(A[256,KTOT](bf16) @ W[KTOT,NCOLS](f32->bf16 in flight) + bias)
//
// 64x64 tile / WG, 8 waves; wave = 1 M-tile x 2 N-tiles, BK=64 -> 4 WMMA/iter.
// Double-buffered LDS with register staging: next W tile's global_load_b128s
// issue before the current tile's compute; one s_barrier per k-iteration.
// All A/B fragments for the iteration are loaded BEFORE the 4 WMMAs so the
// DS waits stagger (dscnt<=6/4/2/0) instead of serializing load->wmma.
// W rows are permuted incrementally (no integer division in the loop).
// ---------------------------------------------------------------------------
template <int NCOLS, int KTOT, bool PERMK, bool RELU_BF16>
__global__ __launch_bounds__(256) void gemm_bf16_wmma(
    const __bf16* __restrict__ A,     // [256, KTOT] bf16 row-major
    const float*  __restrict__ W,     // [KTOT, NCOLS] f32 row-major
    const float*  __restrict__ bias,  // [NCOLS]
    void* __restrict__ outv) {
  constexpr int BM = 64, BN = 64, BK = 64;
  constexpr int BKP = BK + 8;                      // 144B row stride, 16B-aligned
  __shared__ __align__(16) __bf16 Bt[2][BN][BKP];  // [buf][n][k] transposed tile

  const int n0   = blockIdx.x * BN;
  const int m0   = blockIdx.y * BM;
  const int lane = threadIdx.x & 31;
  const int wave = threadIdx.x >> 5;
  const int hl   = lane >> 4;          // half-wave selector (ISA frag layout)
  const int lm   = lane & 15;
  const int mtile  = wave & 3;         // rows m0 + mtile*16 .. +15
  const int ntile0 = (wave >> 2) * 2;  // cols (ntile0+{0,1})*16

  // --- staging: 2 units/thread, unit = rows {2kp, 2kp+1} x cols {c4..c4+3}
  int kp[2], c4[2], cc[2], bb[2];
#pragma unroll
  for (int j = 0; j < 2; ++j) {
    const int u = threadIdx.x + j * 256;  // 0..511
    kp[j] = u >> 4;                       // 0..31
    c4[j] = (u & 15) << 2;                // 0,4,...,60
    cc[j] = 2 * kp[j];                    // krow % FC at kb==0 (BK < FC)
    bb[j] = 0;                            // krow / FC at kb==0
  }

  float4 st0[2], st1[2];
  auto load_tile = [&](int kb_next) {
#pragma unroll
    for (int j = 0; j < 2; ++j) {
      int wrow;
      size_t pair_stride;
      if constexpr (PERMK) {
        wrow = cc[j] * (ROI * ROI) + bb[j];        // c*49 + bin
        pair_stride = (size_t)(ROI * ROI) * NCOLS; // row k+1 -> +49 rows
        cc[j] += BK;
        if (cc[j] >= FC) { cc[j] -= FC; ++bb[j]; } // exact wrap: BK | FC
      } else {
        wrow = kb_next + 2 * kp[j];
        pair_stride = (size_t)NCOLS;
      }
      const float* p = W + (size_t)wrow * NCOLS + n0 + c4[j];
      st0[j] = *(const float4*)(p);
      st1[j] = *(const float4*)(p + pair_stride);
    }
  };

  auto store_tile = [&](int buf) {
#pragma unroll
    for (int j = 0; j < 2; ++j) {
      const int kk = 2 * kp[j];
      v2bf p0; p0.x = (__bf16)st0[j].x; p0.y = (__bf16)st1[j].x;
      v2bf p1; p1.x = (__bf16)st0[j].y; p1.y = (__bf16)st1[j].y;
      v2bf p2; p2.x = (__bf16)st0[j].z; p2.y = (__bf16)st1[j].z;
      v2bf p3; p3.x = (__bf16)st0[j].w; p3.y = (__bf16)st1[j].w;
      *(v2bf*)&Bt[buf][c4[j] + 0][kk] = p0;
      *(v2bf*)&Bt[buf][c4[j] + 1][kk] = p1;
      *(v2bf*)&Bt[buf][c4[j] + 2][kk] = p2;
      *(v2bf*)&Bt[buf][c4[j] + 3][kk] = p3;
    }
  };

  v8f acc0 = {}, acc1 = {};
  const __bf16* abase = A + (size_t)(m0 + mtile * 16 + lm) * KTOT + hl * 8;

  // prologue: stage tile 0
  load_tile(0);
  store_tile(0);
  __syncthreads();

  int cur = 0;
  for (int kb = 0; kb < KTOT; kb += BK) {
    const bool last = (kb + BK >= KTOT);
    if (!last) load_tile(kb + BK);     // global loads issued before compute

    // ---- hoist ALL fragment loads for this iteration ----------------------
    v16bf afrag[2], bfr[2][2];
#pragma unroll
    for (int ks = 0; ks < 2; ++ks) {
      const __bf16* ap = abase + kb + ks * 32;
      const v8bf a_lo = *(const v8bf*)(ap);       // K = hl*8 + 0..7
      const v8bf a_hi = *(const v8bf*)(ap + 16);  // K = 16 + hl*8 + 0..7
      afrag[ks] = __builtin_shufflevector(
          a_lo, a_hi, 0, 1, 2, 3, 4, 5, 6, 7, 8, 9, 10, 11, 12, 13, 14, 15);
#pragma unroll
      for (int nt = 0; nt < 2; ++nt) {
        const __bf16* bp =
            &Bt[cur][(ntile0 + nt) * 16 + lm][ks * 32 + hl * 8];
        const v8bf bl = *(const v8bf*)(bp);
        const v8bf bh = *(const v8bf*)(bp + 16);
        bfr[ks][nt] = __builtin_shufflevector(
            bl, bh, 0, 1, 2, 3, 4, 5, 6, 7, 8, 9, 10, 11, 12, 13, 14, 15);
      }
    }

    // ---- 4 WMMAs, waits can now stagger across the DS queue ---------------
    acc0 = __builtin_amdgcn_wmma_f32_16x16x32_bf16(
        false, afrag[0], false, bfr[0][0], (short)0, acc0, false, false);
    acc1 = __builtin_amdgcn_wmma_f32_16x16x32_bf16(
        false, afrag[0], false, bfr[0][1], (short)0, acc1, false, false);
    acc0 = __builtin_amdgcn_wmma_f32_16x16x32_bf16(
        false, afrag[1], false, bfr[1][0], (short)0, acc0, false, false);
    acc1 = __builtin_amdgcn_wmma_f32_16x16x32_bf16(
        false, afrag[1], false, bfr[1][1], (short)0, acc1, false, false);

    if (!last) {
      store_tile(cur ^ 1);             // write other buffer; reads done above
      __syncthreads();                 // one barrier per k-iteration
    }
    cur ^= 1;
  }

  // ---- epilogue: bias (+ ReLU + bf16 pack for layer 1) --------------------
#pragma unroll
  for (int t = 0; t < 2; ++t) {
    const v8f acc = t ? acc1 : acc0;
    const int col = n0 + (ntile0 + t) * 16 + lm;
    const float bv = bias[col];
#pragma unroll
    for (int r = 0; r < 8; ++r) {
      const int row = m0 + mtile * 16 + hl * 8 + r;
      float v = acc[r] + bv;
      if constexpr (RELU_BF16) {
        v = fmaxf(v, 0.0f);
        ((__bf16*)outv)[(size_t)row * NCOLS + col] = (__bf16)v;
      } else {
        ((float*)outv)[(size_t)row * NCOLS + col] = v;
      }
    }
  }
}

// ---------------------------------------------------------------------------
extern "C" void kernel_launch(void* const* d_in, const int* in_sizes, int n_in,
                              void* d_out, int out_size, void* d_ws,
                              size_t ws_size, hipStream_t stream) {
  (void)in_sizes; (void)n_in; (void)out_size; (void)ws_size;

  const float* fmap  = (const float*)d_in[0];
  const float* boxes = (const float*)d_in[1];
  const float* W1    = (const float*)d_in[2];
  const float* b1    = (const float*)d_in[3];
  const float* W2    = (const float*)d_in[4];
  const float* b2    = (const float*)d_in[5];

  // workspace: A [256, 37632] bf16 (19.27 MB), H [256, 1024] bf16 (0.5 MB)
  __bf16* A = (__bf16*)d_ws;
  __bf16* H = (__bf16*)((char*)d_ws + (size_t)NBOX * K1 * sizeof(__bf16));

  roi_align_pack_kernel<<<dim3(ROI * ROI, NBOX), 256, 0, stream>>>(fmap, boxes, A);

  gemm_bf16_wmma<N1, K1, true, true>
      <<<dim3(N1 / 64, NBOX / 64), 256, 0, stream>>>(A, W1, b1, (void*)H);

  gemm_bf16_wmma<N2, N1, false, false>
      <<<dim3(N2 / 64, NBOX / 64), 256, 0, stream>>>(H, W2, b2, d_out);
}
